// seg_model_11699490915026
// MI455X (gfx1250) — matrix operand order repaired and checked by reading, not compiled, not run on hardware
//
#include <hip/hip_runtime.h>
#include <stdint.h>

// ---------------------------------------------------------------------------
// ChebNet segmentation model on gfx1250 (MI455X).
// All GEMMs (gram, L@X chebyshev chains, T@W, FC head) run through
// v_wmma_f32_16x16x32_bf16 (bf16 inputs, f32 accumulate).  Adjacency and the
// Laplacian are stored bf16 (the dominant 8x2048x2048 traffic term), putting
// the whole pass near the 23.3 TB/s HBM roofline.  Tile staging for the
// contiguous-copy paths uses GLOBAL_LOAD_ASYNC_TO_LDS_B128 (ASYNCcnt) so the
// inner loop never round-trips staged data through VGPRs.
// ---------------------------------------------------------------------------

typedef unsigned short u16;
typedef __attribute__((ext_vector_type(16))) __bf16 v16bf;
typedef __attribute__((ext_vector_type(8)))  float  v8f;

#define BM 128
#define BN 64
#define BK 32
#define SA 40   // ldsA row stride (elements), 80B = multiple of 16B
#define SB 40   // ldsB row stride (elements)
#define NTHREADS 256

__device__ __forceinline__ u16 f2bf(float f) {
  union { float f; uint32_t u; } v; v.f = f;
  uint32_t u = v.u;
  uint32_t r = (u + 0x7FFFu + ((u >> 16) & 1u)) >> 16;
  return (u16)r;
}
__device__ __forceinline__ float bf2f(u16 h) {
  union { uint32_t u; float f; } v; v.u = ((uint32_t)h) << 16; return v.f;
}

union FragU { uint4 u[2]; v16bf v; };

// Async global->LDS copy, 16B per lane (512B per wave-instruction).
// The low 32 bits of a generic pointer to LDS are the LDS byte address
// (aperture rule); VDST supplies that offset.
__device__ __forceinline__ void async_b128(const u16* lds_dst, const u16* gsrc) {
  uint32_t ldsoff = (uint32_t)(uintptr_t)lds_dst;
  asm volatile("global_load_async_to_lds_b128 %0, %1, off"
               :: "v"(ldsoff), "v"(gsrc) : "memory");
}
__device__ __forceinline__ void wait_async0() {
  asm volatile("s_wait_asynccnt 0x0" ::: "memory");
}

// Stage one BMxBK tile of A and one BKxBN tile of B into LDS.
// ldsB is stored transposed: ldsB[j][k], so each lane's B fragment is two
// contiguous 16-byte runs along k.  bTrans==1 means B's source is row-major Nc x K
// (i.e. B = src^T, used for the gram matrix X*X^T) -> contiguous row copy,
// done with async loads.  bTrans==0 performs a transpose scatter (sync path).
__device__ __forceinline__ void stage_tiles(const u16* A, long lda, int m0,
                                            const u16* Bm, long ldb, int n0,
                                            int Nc, int k0, int bTrans,
                                            u16* ldsA, u16* ldsB) {
  const int tid = threadIdx.x;
  { // A: 128 rows x 32 cols, 32 bytes per thread via two async b128
    int row = tid >> 1;
    int cs  = (tid & 1) << 4;
    const u16* src = A + (long)(m0 + row) * lda + k0 + cs;
    u16* dst = ldsA + row * SA + cs;
    async_b128(dst, src);
    async_b128(dst + 8, src + 8);
  }
  if (!bTrans) {           // B row-major K x Nc -> ldsB[j][k] (transpose scatter)
    int k  = tid >> 3;
    int j0 = (tid & 7) << 3;
    const u16* src = Bm + (long)(k0 + k) * ldb + n0 + j0;
    __builtin_prefetch(src + BK * ldb, 0, 1);   // next K-chunk
    u16 tmp[8];
    if (n0 + j0 + 7 < Nc) {
      uint4 d = *(const uint4*)(src);
      const u16* q = (const u16*)&d;
#pragma unroll
      for (int t = 0; t < 8; ++t) tmp[t] = q[t];
    } else {
#pragma unroll
      for (int t = 0; t < 8; ++t) tmp[t] = (n0 + j0 + t < Nc) ? src[t] : (u16)0;
    }
#pragma unroll
    for (int t = 0; t < 8; ++t) ldsB[(j0 + t) * SB + k] = tmp[t];
  } else {                 // B = src^T, src row-major Nc x K: async row copy
    int j   = tid >> 2;
    int seg = (tid & 3) << 3;
    if (n0 + j < Nc) {
      async_b128(ldsB + j * SB + seg, Bm + (long)(n0 + j) * ldb + k0 + seg);
    } else {
      uint4 z = {0, 0, 0, 0};
      *(uint4*)(ldsB + j * SB + seg) = z;
    }
  }
}

// ISA 7.12.2: 16-bit A 16x32 -- lane<16 holds K 0..7,16..23 of row=lane;
// lane>=16 holds K 8..15,24..31 of row=lane-16.  Two 16B LDS reads.
__device__ __forceinline__ v16bf frag_a(const u16* ldsA, int r0, int lane) {
  int g = lane >> 4;
  int row = r0 + (lane & 15);
  const u16* p = ldsA + row * SA + (g << 3);
  FragU f;
  f.u[0] = *(const uint4*)(p);
  f.u[1] = *(const uint4*)(p + 16);
  return f.v;
}
// B 32x16: lane<16 holds K 0..7,16..23; lane>=16 holds K 8..15,24..31;
// col = lane&15. ldsB is [j][k]; two 16B LDS reads.
__device__ __forceinline__ v16bf frag_b(const u16* ldsB, int c0, int lane) {
  int g = lane >> 4;
  int col = c0 + (lane & 15);
  const u16* p = ldsB + col * SB + (g << 3);
  FragU f;
  f.u[0] = *(const uint4*)(p);
  f.u[1] = *(const uint4*)(p + 16);
  return f.v;
}

__device__ __forceinline__ void gemm_core(const u16* A, long lda,
                                          const u16* Bm, long ldb, int bTrans,
                                          int m0, int n0, int Nc, int K,
                                          u16* ldsA, u16* ldsB, v8f acc[2][2]) {
  const int lane = threadIdx.x & 31;
  const int w  = threadIdx.x >> 5;
  const int wm = w >> 1, wn = w & 1;   // 4x2 wave grid over 128x64
  for (int k0 = 0; k0 < K; k0 += BK) {
    __syncthreads();   // all waves' fragment reads of previous chunk done
    stage_tiles(A, lda, m0, Bm, ldb, n0, Nc, k0, bTrans, ldsA, ldsB);
    wait_async0();     // our async LDS writes landed
    __syncthreads();   // everyone's writes visible
    v16bf a0 = frag_a(ldsA, wm * 32,      lane);
    v16bf a1 = frag_a(ldsA, wm * 32 + 16, lane);
    v16bf b0 = frag_b(ldsB, wn * 32,      lane);
    v16bf b1 = frag_b(ldsB, wn * 32 + 16, lane);
    acc[0][0] = __builtin_amdgcn_wmma_f32_16x16x32_bf16(false, a0, false, b0, (short)0, acc[0][0], false, false);
    acc[0][1] = __builtin_amdgcn_wmma_f32_16x16x32_bf16(false, a0, false, b1, (short)0, acc[0][1], false, false);
    acc[1][0] = __builtin_amdgcn_wmma_f32_16x16x32_bf16(false, a1, false, b0, (short)0, acc[1][0], false, false);
    acc[1][1] = __builtin_amdgcn_wmma_f32_16x16x32_bf16(false, a1, false, b1, (short)0, acc[1][1], false, false);
  }
}

// Configurable-epilogue GEMM:
//   v = alpha*acc [+ addScale*addP] [+ bias[col]] [+ br[col]]; optional relu;
//   optional f32 store, optional bf16 store.  blockIdx.z = batch.
__global__ __launch_bounds__(NTHREADS)
void gemm_epi(const u16* __restrict__ A, long lda, long bsA,
              const u16* __restrict__ Bm, long ldb, long bsB, int bTrans,
              int M, int Nc, int K, float alpha,
              const float* __restrict__ addP, long ldadd, long bsAdd, float addScale,
              const float* __restrict__ bias, const float* __restrict__ brv, int relu,
              float* __restrict__ outF, long ldo, long bsO,
              u16* __restrict__ outB, long ldob, long bsOb) {
  __shared__ u16 ldsA[BM * SA];
  __shared__ u16 ldsB[BN * SB];
  const int z = blockIdx.z;
  const u16* Az = A + (long)z * bsA;
  const u16* Bz = Bm + (long)z * bsB;
  const int m0 = blockIdx.y * BM;
  const int n0 = blockIdx.x * BN;
  v8f acc[2][2] = {};
  gemm_core(Az, lda, Bz, ldb, bTrans, m0, n0, Nc, K, ldsA, ldsB, acc);
  const int lane = threadIdx.x & 31;
  const int w = threadIdx.x >> 5;
  const int wm = w >> 1, wn = w & 1;
  const int g = lane >> 4, ln = lane & 15;
  const float* addPz = addP ? addP + (long)z * bsAdd : nullptr;
  float* outFz = outF ? outF + (long)z * bsO : nullptr;
  u16* outBz = outB ? outB + (long)z * bsOb : nullptr;
#pragma unroll
  for (int mt = 0; mt < 2; ++mt)
#pragma unroll
    for (int nt = 0; nt < 2; ++nt)
#pragma unroll
      for (int r = 0; r < 8; ++r) {
        int row = m0 + wm * 32 + mt * 16 + r + (g << 3);
        int col = n0 + wn * 32 + nt * 16 + ln;
        if (row < M && col < Nc) {
          float v = alpha * acc[mt][nt][r];
          if (addPz) v += addScale * addPz[(long)row * ldadd + col];
          if (bias)  v += bias[col];
          if (brv)   v += brv[col];
          if (relu)  v = fmaxf(v, 0.f);
          if (outFz) outFz[(long)row * ldo + col] = v;
          if (outBz) outBz[(long)row * ldob + col] = f2bf(v);
        }
      }
}

// Gram + gaussian kernel: adj = exp(2*(X.X^T) - sq_i - sq_j), bf16 output.
__global__ __launch_bounds__(NTHREADS)
void gemm_gram(const u16* __restrict__ X, long lda, int Nrow, int K,
               const float* __restrict__ sq, u16* __restrict__ adjOut) {
  __shared__ u16 ldsA[BM * SA];
  __shared__ u16 ldsB[BN * SB];
  const int z = blockIdx.z;
  const u16* Xz = X + (long)z * Nrow * lda;
  const float* sqz = sq + (long)z * Nrow;
  u16* adj = adjOut + (long)z * Nrow * Nrow;
  const int m0 = blockIdx.y * BM;
  const int n0 = blockIdx.x * BN;
  v8f acc[2][2] = {};
  gemm_core(Xz, lda, Xz, lda, /*bTrans=*/1, m0, n0, Nrow, K, ldsA, ldsB, acc);
  const int lane = threadIdx.x & 31;
  const int w = threadIdx.x >> 5;
  const int wm = w >> 1, wn = w & 1;
  const int g = lane >> 4, ln = lane & 15;
#pragma unroll
  for (int mt = 0; mt < 2; ++mt)
#pragma unroll
    for (int nt = 0; nt < 2; ++nt)
#pragma unroll
      for (int r = 0; r < 8; ++r) {
        int row = m0 + wm * 32 + mt * 16 + r + (g << 3);
        int col = n0 + wn * 32 + nt * 16 + ln;
        if (row < Nrow && col < Nrow) {
          float v = __expf(2.f * acc[mt][nt][r] - sqz[row] - sqz[col]);
          adj[(long)row * Nrow + col] = f2bf(v);
        }
      }
}

// ------------------------------ small kernels ------------------------------

__global__ void k_convert(u16* dst, const float* src, long n) {
  long i = (long)blockIdx.x * blockDim.x + threadIdx.x;
  long stride = (long)gridDim.x * blockDim.x;
  for (; i < n; i += stride) dst[i] = f2bf(src[i]);
}

__global__ void k_pad_convert(u16* dst, const float* src, int srcR, int srcC,
                              int dstR, int dstC) {
  int i = blockIdx.x * blockDim.x + threadIdx.x;
  int total = dstR * dstC;
  if (i >= total) return;
  int r = i / dstC, c = i - r * dstC;
  float v = (r < srcR && c < srcC) ? src[r * srcC + c] : 0.f;
  dst[i] = f2bf(v);
}

__global__ void k_build_input(const float* x, const int* cat, float* X0f,
                              u16* X0bf, u16* xgbf, int rows, int Np) {
  int idx = blockIdx.x * blockDim.x + threadIdx.x;
  if (idx >= rows) return;
  int b = idx / Np;
  int c = cat[b];
  const float* xp = x + (long)idx * 6;
  float* f  = X0f  + (long)idx * 32;
  u16*   fb = X0bf + (long)idx * 32;
  u16*   gb = xgbf + (long)idx * 32;
  for (int d = 0; d < 32; ++d) {
    float v = (d < 6) ? xp[d] : ((d < 22) ? ((d - 6) == c ? 1.f : 0.f) : 0.f);
    f[d] = v;
    fb[d] = f2bf(v);
    gb[d] = (d < 6) ? f2bf(xp[d]) : (u16)0;
  }
}

__global__ void k_rowsq(const float* X, int stride, int D, float* sq, int rows) {
  int r = blockIdx.x * blockDim.x + threadIdx.x;
  if (r >= rows) return;
  const float* p = X + (long)r * stride;
  float s = 0.f;
  for (int d = 0; d < D; ++d) { float v = p[d]; s += v * v; }
  sq[r] = s;
}

__global__ void k_rowsum_rinv(const u16* adj, int Ncols, float* rinv) {
  __shared__ float red[256];
  long row = blockIdx.x;
  const u16* p = adj + row * Ncols;
  float s = 0.f;
  for (int j = threadIdx.x; j < Ncols; j += 256) s += bf2f(p[j]);
  red[threadIdx.x] = s;
  __syncthreads();
  for (int st = 128; st > 0; st >>= 1) {
    if (threadIdx.x < st) red[threadIdx.x] += red[threadIdx.x + st];
    __syncthreads();
  }
  if (threadIdx.x == 0) rinv[row] = rsqrtf(red[0]);
}

__global__ void k_make_L(const u16* adj, const float* rinv, u16* L, int Np) {
  int z = blockIdx.z;
  int i = blockIdx.y;
  int j = blockIdx.x * blockDim.x + threadIdx.x;
  long idx = ((long)z * Np + i) * Np + j;
  float a = bf2f(adj[idx]) * rinv[(long)z * Np + i] * rinv[(long)z * Np + j];
  float v = (i == j ? 1.f : 0.f) - a;
  L[idx] = f2bf(v);
}

__global__ void k_init_bias(float* out, const float* bias, int F, long total) {
  long i = (long)blockIdx.x * blockDim.x + threadIdx.x;
  long stride = (long)gridDim.x * blockDim.x;
  for (; i < total; i += stride) out[i] = bias[(int)(i % F)];
}

__global__ void k_relu_feat(const float* src, const float* brv, int F, long total,
                            float* dstF, u16* dstB) {
  long i = (long)blockIdx.x * blockDim.x + threadIdx.x;
  long stride = (long)gridDim.x * blockDim.x;
  for (; i < total; i += stride) {
    int col = (int)(i % F);
    float v = fmaxf(src[i] + brv[col], 0.f);
    if (dstF) dstF[i] = v;
    dstB[i] = f2bf(v);
  }
}

// ------------------------------ host driver --------------------------------

extern "C" void kernel_launch(void* const* d_in, const int* in_sizes, int n_in,
                              void* d_out, int out_size, void* d_ws, size_t ws_size,
                              hipStream_t stream) {
  (void)in_sizes; (void)n_in; (void)out_size; (void)ws_size;
  const int Bn = 8, Np = 2048;
  const long MF = (long)Bn * Np;  // 16384 flattened rows

  const float* x   = (const float*)d_in[0];
  const int*   cat = (const int*)d_in[1];
  const float* W0  = (const float*)d_in[2];
  const float* b0  = (const float*)d_in[3];
  const float* W1  = (const float*)d_in[4];
  const float* b1  = (const float*)d_in[5];
  const float* W2  = (const float*)d_in[6];
  const float* b2  = (const float*)d_in[7];
  const float* fw0 = (const float*)d_in[8];
  const float* fb0 = (const float*)d_in[9];
  const float* fw1 = (const float*)d_in[10];
  const float* fb1 = (const float*)d_in[11];
  const float* fw2 = (const float*)d_in[12];
  const float* fb2 = (const float*)d_in[13];
  const float* br0 = (const float*)d_in[14];
  const float* br1 = (const float*)d_in[15];
  const float* br2 = (const float*)d_in[16];
  const float* br3 = (const float*)d_in[17];
  const float* br4 = (const float*)d_in[18];
  const float* br5 = (const float*)d_in[19];
  float* outp = (float*)d_out;

  char* base = (char*)d_ws;
  size_t off = 0;
  auto carve = [&](size_t bytes) -> char* {
    char* p = base + off;
    off += (bytes + 255) & ~(size_t)255;
    return p;
  };
  u16*   adjbf = (u16*)carve((size_t)Bn * Np * Np * 2);
  u16*   Lbf   = (u16*)carve((size_t)Bn * Np * Np * 2);
  float* rinv  = (float*)carve((size_t)MF * 4);
  float* sqv   = (float*)carve((size_t)MF * 4);
  float* X0f   = (float*)carve((size_t)MF * 32 * 4);
  u16*   X0bf  = (u16*)carve((size_t)MF * 32 * 2);
  u16*   xgbf  = (u16*)carve((size_t)MF * 32 * 2);
  float* tf[3]; u16* tb[3];
  for (int i = 0; i < 3; ++i) {
    tf[i] = (float*)carve((size_t)MF * 512 * 4);
    tb[i] = (u16*)carve((size_t)MF * 512 * 2);
  }
  float* out_acc  = (float*)carve((size_t)MF * 1024 * 4);
  u16*   prebf    = (u16*)carve((size_t)MF * 1024 * 2);
  float* featf_a  = (float*)carve((size_t)MF * 128 * 4);
  u16*   featbf_a = (u16*)carve((size_t)MF * 128 * 2);
  float* featf_b  = (float*)carve((size_t)MF * 512 * 4);
  u16*   featbf_b = (u16*)carve((size_t)MF * 512 * 2);   // = x1_skip (bf16)
  u16*   fc0bf    = (u16*)carve((size_t)MF * 512 * 2);
  float* fctmp    = (float*)carve((size_t)MF * 128 * 4);
  u16*   fc1bf    = (u16*)carve((size_t)MF * 128 * 2);
  u16* w0bf  = (u16*)carve((size_t)6 * 32 * 128 * 2);
  u16* w1bf  = (u16*)carve((size_t)5 * 128 * 512 * 2);
  u16* w2bf  = (u16*)carve((size_t)3 * 512 * 1024 * 2);
  u16* fw0bf = (u16*)carve((size_t)1024 * 512 * 2);
  u16* fw1bf = (u16*)carve((size_t)1024 * 128 * 2);
  u16* fw2bf = (u16*)carve((size_t)128 * 64 * 2);

  auto gemm = [&](const u16* A, long lda, long bsA, const u16* Bm, long ldb, long bsB,
                  int M, int Nc, int K, float alpha,
                  const float* addP, long ldadd, long bsAdd, float addScale,
                  const float* bias, const float* brv, int relu,
                  float* outF, long ldo, long bsO,
                  u16* outB, long ldob, long bsOb, int batches) {
    dim3 grid((Nc + BN - 1) / BN, (M + BM - 1) / BM, batches);
    gemm_epi<<<grid, NTHREADS, 0, stream>>>(A, lda, bsA, Bm, ldb, bsB, 0, M, Nc, K,
                                            alpha, addP, ldadd, bsAdd, addScale,
                                            bias, brv, relu, outF, ldo, bsO,
                                            outB, ldob, bsOb);
  };

  auto lap = [&](const float* Xf, int stride, int D, const u16* Xbf, int ldaBf, int Kd) {
    int rows = (int)MF;
    k_rowsq<<<(rows + 255) / 256, 256, 0, stream>>>(Xf, stride, D, sqv, rows);
    dim3 g(Np / BN, Np / BM, Bn);
    gemm_gram<<<g, NTHREADS, 0, stream>>>(Xbf, ldaBf, Np, Kd, sqv, adjbf);
    k_rowsum_rinv<<<rows, 256, 0, stream>>>(adjbf, Np, rinv);
    dim3 gl(Np / 256, Np, Bn);
    k_make_L<<<gl, 256, 0, stream>>>(adjbf, rinv, Lbf, Np);
  };

  auto conv = [&](const u16* Xbf, const float* Xf, int Dp, int F, int Kc,
                  const u16* Wb, const float* bias) {
    long totalF = MF * F;
    int blk = (int)((totalF + 255) / 256); if (blk > 16384) blk = 16384;
    k_init_bias<<<blk, 256, 0, stream>>>(out_acc, bias, F, totalF);
    // k = 0: out += X0 @ W0
    gemm(Xbf, Dp, 0, Wb, F, 0, (int)MF, F, Dp, 1.f,
         out_acc, F, 0, 1.f, nullptr, nullptr, 0, out_acc, F, 0, nullptr, 0, 0, 1);
    // T1 = L @ X0
    gemm(Lbf, Np, (long)Np * Np, Xbf, Dp, (long)Np * Dp, Np, Dp, Np, 1.f,
         nullptr, 0, 0, 0.f, nullptr, nullptr, 0,
         tf[0], Dp, (long)Np * Dp, tb[0], Dp, (long)Np * Dp, Bn);
    // out += T1 @ W1
    gemm(tb[0], Dp, 0, Wb + (long)Dp * F, F, 0, (int)MF, F, Dp, 1.f,
         out_acc, F, 0, 1.f, nullptr, nullptr, 0, out_acc, F, 0, nullptr, 0, 0, 1);
    const float* p0f = Xf;
    const float* p1f = tf[0];
    const u16*   p1b = tb[0];
    for (int k = 2; k < Kc; ++k) {
      int s = (k - 1) % 3;
      float* nf = tf[s]; u16* nb = tb[s];
      // Tk = 2*L@T(k-1) - T(k-2)
      gemm(Lbf, Np, (long)Np * Np, p1b, Dp, (long)Np * Dp, Np, Dp, Np, 2.f,
           p0f, Dp, (long)Np * Dp, -1.f, nullptr, nullptr, 0,
           nf, Dp, (long)Np * Dp, nb, Dp, (long)Np * Dp, Bn);
      // out += Tk @ Wk
      gemm(nb, Dp, 0, Wb + (long)k * Dp * F, F, 0, (int)MF, F, Dp, 1.f,
           out_acc, F, 0, 1.f, nullptr, nullptr, 0, out_acc, F, 0, nullptr, 0, 0, 1);
      p0f = p1f; p1f = nf; p1b = nb;
    }
  };

  // ---- weight conversion (bf16, with zero padding where shapes need it) ----
  for (int k = 0; k < 6; ++k)
    k_pad_convert<<<(32 * 128 + 255) / 256, 256, 0, stream>>>(
        w0bf + k * 32 * 128, W0 + (long)k * 22 * 128, 22, 128, 32, 128);
  k_convert<<<4096, 256, 0, stream>>>(w1bf, W1, (long)5 * 128 * 512);
  k_convert<<<8192, 256, 0, stream>>>(w2bf, W2, (long)3 * 512 * 1024);
  k_convert<<<4096, 256, 0, stream>>>(fw0bf, fw0, (long)1024 * 512);
  k_convert<<<1024, 256, 0, stream>>>(fw1bf, fw1, (long)1024 * 128);
  k_pad_convert<<<(128 * 64 + 255) / 256, 256, 0, stream>>>(fw2bf, fw2, 128, 50, 128, 64);

  // ---- input features: [x | one_hot(cat,16)] padded to 32 ----
  k_build_input<<<((int)MF + 255) / 256, 256, 0, stream>>>(x, cat, X0f, X0bf, xgbf,
                                                           (int)MF, Np);

  // ---- Laplacian #1 from geometry (D=6, padded gram K=32) ----
  lap(x, 6, 6, xgbf, 32, 32);

  // ---- conv0: cin=22(pad32) -> 128, K=6 ----
  conv(X0bf, X0f, 32, 128, 6, w0bf, b0);

  // ---- Laplacian #2 from pre-relu conv0 output (D=128) ----
  k_convert<<<8192, 256, 0, stream>>>(prebf, out_acc, MF * 128);
  lap(out_acc, 128, 128, prebf, 128, 128);
  k_relu_feat<<<8192, 256, 0, stream>>>(out_acc, br0, 128, MF * 128, featf_a, featbf_a);

  // ---- conv1: 128 -> 512, K=5 ----
  conv(featbf_a, featf_a, 128, 512, 5, w1bf, b1);

  // ---- Laplacian #3 from pre-relu conv1 output (D=512) ----
  k_convert<<<8192, 256, 0, stream>>>(prebf, out_acc, MF * 512);
  lap(out_acc, 512, 512, prebf, 512, 512);
  k_relu_feat<<<8192, 256, 0, stream>>>(out_acc, br1, 512, MF * 512, featf_b, featbf_b);

  // ---- conv2: 512 -> 1024, K=3 (last Laplacian recompute is unused) ----
  conv(featbf_b, featf_b, 512, 1024, 3, w2bf, b2);
  k_relu_feat<<<8192, 256, 0, stream>>>(out_acc, br2, 1024, MF * 1024, nullptr, prebf);

  // ---- FC head ----
  // fc0: 1024 -> 512, relu(acc + fb0 + br3)
  gemm(prebf, 1024, 0, fw0bf, 512, 0, (int)MF, 512, 1024, 1.f,
       nullptr, 0, 0, 0.f, fb0, br3, 1, nullptr, 0, 0, fc0bf, 512, 0, 1);
  // fc1: concat(out, x1_skip) @ fw1 split into two K=512 GEMMs
  gemm(fc0bf, 512, 0, fw1bf, 128, 0, (int)MF, 128, 512, 1.f,
       nullptr, 0, 0, 0.f, nullptr, nullptr, 0, fctmp, 128, 0, nullptr, 0, 0, 1);
  gemm(featbf_b, 512, 0, fw1bf + (long)512 * 128, 128, 0, (int)MF, 128, 512, 1.f,
       fctmp, 128, 0, 1.f, fb1, br4, 1, nullptr, 0, 0, fc1bf, 128, 0, 1);
  // fc2: 128 -> 50 (weights padded to 64 cols), relu(acc + fb2 + br5) -> d_out
  gemm(fc1bf, 128, 0, fw2bf, 64, 0, (int)MF, 50, 128, 1.f,
       nullptr, 0, 0, 0.f, fb2, br5, 1, outp, 50, 0, nullptr, 0, 0, 1);
}
